// CrossAnalysisModel_60318520705769
// MI455X (gfx1250) — matrix operand order repaired and checked by reading, not compiled, not run on hardware
//
#include <hip/hip_runtime.h>
#include <hip/hip_bf16.h>
#include <cstdint>
#include <cstddef>

// ---------------------------------------------------------------------------
// CrossAnalysisModel (2-layer hetero GATv2) for MI455X / gfx1250 (wave32).
//
// Bandwidth-first design (MI455X: 23.3 TB/s HBM; GEMM FLOPs trivial vs WMMA):
//  * Activations stored FP16 -> feed v_wmma_f32_16x16x32_f16, fp32 accumulate.
//  * Weights pre-packed into per-lane WMMA B-fragment order so the GEMM inner
//    loop does only global_load_b128s (no strided u16 loads).
//  * Edge phase wave-per-edge: 8 contiguous floats/lane -> coalesced b128
//    gathers and coalesced global_atomic_add_f32 scatters.
//  * Segment softmax: atomic-max (CAS), exp + atomic-add, scatter.
//
// Input index map (setup_inputs dict order; params flattened jax.tree-style,
// i.e. sorted dict keys at every level):
//   0 x_user 1 x_poi 2 x_cate 3 ea_up 4 ea_pu 5 ea_pc 6 ea_cp
//   7 src_up 8 dst_up 9 src_pu 10 dst_pu 11 src_pc 12 dst_pc 13 src_cp 14 dst_cp
//   convs sorted {cp,pc,pu,up} x {c1,c2} x sorted {We,Wl,Wr,att,bias,bl,br}:
//     cp.c1 @15, cp.c2 @22, pc.c1 @29, pc.c2 @36, pu.c1 @43, pu.c2 @50,
//     up.c1 @57, up.c2 @64
//   enc (sorted {W,b,beta,g}): enc_cate @71, enc_poi @75, enc_user @79
// ---------------------------------------------------------------------------

typedef __attribute__((ext_vector_type(16))) _Float16 v16h;
typedef __attribute__((ext_vector_type(8)))  float    v8f;

#define N_USER 50000
#define N_POI  20000
#define N_CATE 1000
#define E_UP   250000
#define E_PC   40000

static inline int ceil_div(int a, int b) { return (a + b - 1) / b; }

// -------------------------------- device helpers ---------------------------

__device__ __forceinline__ void atomicMaxF(float* addr, float val) {
  unsigned int* ia = reinterpret_cast<unsigned int*>(addr);
  unsigned int old = __float_as_uint(*addr);
  while (__uint_as_float(old) < val) {
    unsigned int assumed = old;
    old = atomicCAS(ia, assumed, __float_as_uint(val));
    if (old == assumed) break;
  }
}

// -------------------------------- kernels ----------------------------------

__global__ void fill_f32(float* __restrict__ p, float v, int n) {
  int i = blockIdx.x * blockDim.x + threadIdx.x;
  if (i < n) p[i] = v;
}

// Pack W[K,D] (f32, row-major) into WMMA B-fragment order, f16:
//   Wp[((tn*(K/32) + k32)*32 + lane)*16 + j] = W[k32*32 + (j>>3)*16 + (lane>>4)*8 + (j&7)][tn*16 + (lane&15)]
// so each lane's 16 halves for one (tile_n, k-step) are contiguous (32B).
__global__ void pack_w_f16(const float* __restrict__ W, _Float16* __restrict__ Wp,
                           int K, int D) {
  int i = blockIdx.x * blockDim.x + threadIdx.x;
  if (i >= K * D) return;
  int j    = i & 15;
  int lane = (i >> 4) & 31;
  int rest = i >> 9;                 // tn*(K/32) + k32
  int ksteps = K >> 5;
  int k32 = rest % ksteps;
  int tn  = rest / ksteps;
  int k = (k32 << 5) + ((j >> 3) << 4) + ((lane >> 4) << 3) + (j & 7);
  int n = (tn << 4) + (lane & 15);
  Wp[i] = (_Float16)W[(size_t)k * D + n];
}

// encoder: Linear -> eval-BatchNorm -> ReLU, output fp16 (only feeds WMMA GEMMs)
__global__ void encode_kernel(const float* __restrict__ x, const float* __restrict__ W,
                              const float* __restrict__ b, const float* __restrict__ g,
                              const float* __restrict__ beta, _Float16* __restrict__ out,
                              int N, int F) {
  int i = blockIdx.x * blockDim.x + threadIdx.x;
  if (i >= N * 64) return;
  int n = i >> 6, j = i & 63;
  float acc = b[j];
  for (int k = 0; k < F; ++k) acc += x[n * F + k] * W[k * 64 + j];
  float h = g[j] * acc * 0.999995000037499678f + beta[j];  // 1/sqrt(1+1e-5)
  out[i] = (_Float16)(h > 0.f ? h : 0.f);
}

// Y[N,D] = X[N,K](f16) @ W(packed f16 fragments) + bias[D], fp32 out.
// K%32==0, D%16==0. One wave per 16x16 tile, 8 waves/block.
// Fragment layouts per CDNA5 ISA 7.12.2.
__global__ void wmma_gemm_bias(const _Float16* __restrict__ X,
                               const _Float16* __restrict__ Wp,
                               const float* __restrict__ bias,
                               float* __restrict__ Y,
                               int N, int K, int D) {
  const int lane = threadIdx.x & 31;
  const int wid  = threadIdx.x >> 5;
  const int tilesD = D >> 4;
  const int tilesN = (N + 15) >> 4;
  const int tile = blockIdx.x * (blockDim.x >> 5) + wid;
  if (tile >= tilesN * tilesD) return;      // wave-uniform: EXEC stays all-ones
  const int tm = tile / tilesD;
  const int tn = tile - tm * tilesD;

  const int mrow  = tm * 16 + (lane & 15);
  const int arow  = (mrow < N) ? mrow : 0;  // clamp loads; stores guarded below
  const int akoff = (lane >> 4) << 3;       // lanes 16-31: K base +8
  const int ncol  = tn * 16 + (lane & 15);
  const int ksteps = K >> 5;

  const _Float16* Xrow = X + (size_t)arow * K + akoff;
  const _Float16* Wfrag = Wp + (((size_t)tn * ksteps) << 9) + ((size_t)lane << 4);

  v8f acc = {};
  for (int k32 = 0; k32 < ksteps; ++k32) {
    v16h a;
    const _Float16* Xr = Xrow + (k32 << 5);
#pragma unroll
    for (int i = 0; i < 8; ++i) a[i] = Xr[i];           // -> global_load_b128 x2
#pragma unroll
    for (int i = 0; i < 8; ++i) a[8 + i] = Xr[16 + i];
    v16h b = *(const v16h*)(Wfrag + ((size_t)k32 << 9)); // contiguous 32B/lane
    acc = __builtin_amdgcn_wmma_f32_16x16x32_f16(false, a, false, b,
                                                 (short)0, acc, false, false);
  }
  const float bj = bias[ncol];
  const int mbase = tm * 16 + ((lane >> 4) << 3);
#pragma unroll
  for (int r = 0; r < 8; ++r) {
    int m = mbase + r;
    if (m < N) Y[(size_t)m * D + ncol] = acc[r] + bj;
  }
}

// logits: wave per edge. e = leaky_relu(xl[src]+xr[dst]+ea*We, 0.2);
// logit[e,h] = dot(e[h], att[h]).  d = H*64, R = d/32 contiguous elems/lane.
template <int H>
__global__ void gat_logit(const float* __restrict__ xl, const float* __restrict__ xr,
                          const float* __restrict__ We, const float* __restrict__ att,
                          const float* __restrict__ ea, const int* __restrict__ src,
                          const int* __restrict__ dst, float* __restrict__ logit, int E) {
  const int lane = threadIdx.x & 31;
  const int e = blockIdx.x * (blockDim.x >> 5) + (threadIdx.x >> 5);
  if (e >= E) return;
  constexpr int d = H * 64;
  constexpr int R = d / 32;
  const int s = src[e], t = dst[e];
  const float w = ea[e];
  const float* xls = xl + (size_t)s * d;
  const float* xrt = xr + (size_t)t * d;
  float acc = 0.f;
#pragma unroll
  for (int r = 0; r < R; ++r) {
    int idx = lane * R + r;                 // head = idx>>6, att is flat [H*64]
    float v = xls[idx] + xrt[idx] + w * We[idx];
    v = (v > 0.f) ? v : 0.2f * v;
    acc += v * att[idx];
  }
  constexpr int group = 64 / R;             // lanes covering one head
#pragma unroll
  for (int off = group >> 1; off > 0; off >>= 1)
    acc += __shfl_xor(acc, off);
  if ((lane & (group - 1)) == 0)
    logit[(size_t)e * H + (lane / group)] = acc;
}

__global__ void seg_max_atomic(const float* __restrict__ logit, const int* __restrict__ dst,
                               float* __restrict__ m, int E, int H) {
  int i = blockIdx.x * blockDim.x + threadIdx.x;
  if (i >= E * H) return;
  int e = i / H, h = i - e * H;
  atomicMaxF(&m[(size_t)dst[e] * H + h], logit[i]);
}

__global__ void seg_expsum(float* __restrict__ logit, const int* __restrict__ dst,
                           const float* __restrict__ m, float* __restrict__ denom,
                           int E, int H) {
  int i = blockIdx.x * blockDim.x + threadIdx.x;
  if (i >= E * H) return;
  int e = i / H, h = i - e * H;
  size_t di = (size_t)dst[e] * H + h;
  float a = expf(logit[i] - m[di]);
  logit[i] = a;                              // overwrite logit with exp-numerator
  atomicAdd(&denom[di], a);
}

// scatter: wave per edge; agg[dst] += xl[src] * alpha
template <int H>
__global__ void gat_scatter(const float* __restrict__ xl, const float* __restrict__ anum,
                            const float* __restrict__ denom, const int* __restrict__ src,
                            const int* __restrict__ dst, float* __restrict__ agg, int E) {
  const int lane = threadIdx.x & 31;
  const int e = blockIdx.x * (blockDim.x >> 5) + (threadIdx.x >> 5);
  if (e >= E) return;
  constexpr int d = H * 64;
  constexpr int R = d / 32;
  const int s = src[e], t = dst[e];
  const float* xls = xl + (size_t)s * d;
  float* aggd = agg + (size_t)t * d;
#pragma unroll
  for (int r = 0; r < R; ++r) {
    int idx = lane * R + r;
    int h = idx >> 6;
    float al = anum[(size_t)e * H + h] / (denom[(size_t)t * H + h] + 1e-16f);
    atomicAdd(&aggd[idx], xls[idx] * al);
  }
}

// layer-1 finalize (+bias, ReLU, -> fp16 for next layer's WMMA)
__global__ void finalize_relu_h(const float* __restrict__ agg, const float* __restrict__ bias,
                                _Float16* __restrict__ out, int N, int d) {
  int i = blockIdx.x * blockDim.x + threadIdx.x;
  if (i >= N * d) return;
  float v = agg[i] + bias[i % d];
  out[i] = (_Float16)(v > 0.f ? v : 0.f);
}

__global__ void finalize_relu_pair_h(const float* __restrict__ aggA, const float* __restrict__ biasA,
                                     const float* __restrict__ aggB, const float* __restrict__ biasB,
                                     _Float16* __restrict__ out, int N, int d) {
  int i = blockIdx.x * blockDim.x + threadIdx.x;
  if (i >= N * d) return;
  int j = i % d;
  float v = 0.5f * (aggA[i] + biasA[j] + aggB[i] + biasB[j]);
  out[i] = (_Float16)(v > 0.f ? v : 0.f);
}

// layer-2 finalize (+bias, fp32 to d_out)
__global__ void out_single(const float* __restrict__ agg, const float* __restrict__ bias,
                           float* __restrict__ out, int N, int d) {
  int i = blockIdx.x * blockDim.x + threadIdx.x;
  if (i >= N * d) return;
  out[i] = agg[i] + bias[i % d];
}

__global__ void out_pair(const float* __restrict__ aggA, const float* __restrict__ biasA,
                         const float* __restrict__ aggB, const float* __restrict__ biasB,
                         float* __restrict__ out, int N, int d) {
  int i = blockIdx.x * blockDim.x + threadIdx.x;
  if (i >= N * d) return;
  int j = i % d;
  out[i] = 0.5f * (aggA[i] + biasA[j] + aggB[i] + biasB[j]);
}

// ------------------------------ host orchestration -------------------------

struct ConvBufs {
  _Float16 *w16a, *w16b;
  float *xl, *xr, *logit, *segm, *denom;
};

static void run_conv(hipStream_t stream, const ConvBufs& B,
                     const _Float16* Xsrc, int Ns, const _Float16* Xdst, int Nd,
                     int K, int H,
                     const float* We, const float* Wl, const float* Wr,
                     const float* att, const float* bl, const float* br,
                     const float* ea, const int* src, const int* dst, int E,
                     float* agg) {
  const int d = H * 64;
  pack_w_f16<<<ceil_div(K * d, 256), 256, 0, stream>>>(Wl, B.w16a, K, d);
  pack_w_f16<<<ceil_div(K * d, 256), 256, 0, stream>>>(Wr, B.w16b, K, d);
  {
    int tiles = ceil_div(Ns, 16) * (d / 16);
    wmma_gemm_bias<<<ceil_div(tiles, 8), 256, 0, stream>>>(Xsrc, B.w16a, bl, B.xl, Ns, K, d);
  }
  {
    int tiles = ceil_div(Nd, 16) * (d / 16);
    wmma_gemm_bias<<<ceil_div(tiles, 8), 256, 0, stream>>>(Xdst, B.w16b, br, B.xr, Nd, K, d);
  }
  fill_f32<<<ceil_div(Nd * H, 256), 256, 0, stream>>>(B.segm, -3.0e38f, Nd * H);
  fill_f32<<<ceil_div(Nd * H, 256), 256, 0, stream>>>(B.denom, 0.f, Nd * H);
  fill_f32<<<ceil_div(Nd * d, 256), 256, 0, stream>>>(agg, 0.f, Nd * d);
  if (H == 4)
    gat_logit<4><<<ceil_div(E, 8), 256, 0, stream>>>(B.xl, B.xr, We, att, ea, src, dst, B.logit, E);
  else
    gat_logit<1><<<ceil_div(E, 8), 256, 0, stream>>>(B.xl, B.xr, We, att, ea, src, dst, B.logit, E);
  seg_max_atomic<<<ceil_div(E * H, 256), 256, 0, stream>>>(B.logit, dst, B.segm, E, H);
  seg_expsum<<<ceil_div(E * H, 256), 256, 0, stream>>>(B.logit, dst, B.segm, B.denom, E, H);
  if (H == 4)
    gat_scatter<4><<<ceil_div(E, 8), 256, 0, stream>>>(B.xl, B.logit, B.denom, src, dst, agg, E);
  else
    gat_scatter<1><<<ceil_div(E, 8), 256, 0, stream>>>(B.xl, B.logit, B.denom, src, dst, agg, E);
}

extern "C" void kernel_launch(void* const* d_in, const int* in_sizes, int n_in,
                              void* d_out, int out_size, void* d_ws, size_t ws_size,
                              hipStream_t stream) {
  (void)in_sizes; (void)n_in; (void)out_size; (void)ws_size;

  const float* x_user = (const float*)d_in[0];
  const float* x_poi  = (const float*)d_in[1];
  const float* x_cate = (const float*)d_in[2];
  const float* ea_up  = (const float*)d_in[3];
  const float* ea_pu  = (const float*)d_in[4];
  const float* ea_pc  = (const float*)d_in[5];
  const float* ea_cp  = (const float*)d_in[6];
  const int* src_up = (const int*)d_in[7];
  const int* dst_up = (const int*)d_in[8];
  const int* src_pu = (const int*)d_in[9];
  const int* dst_pu = (const int*)d_in[10];
  const int* src_pc = (const int*)d_in[11];
  const int* dst_pc = (const int*)d_in[12];
  const int* src_cp = (const int*)d_in[13];
  const int* dst_cp = (const int*)d_in[14];

  // conv param bases: fields We+0, Wl+1, Wr+2, att+3, bias+4, bl+5, br+6
  const int CP1 = 15, CP2 = 22, PC1 = 29, PC2 = 36, PU1 = 43, PU2 = 50, UP1 = 57, UP2 = 64;
  auto F = [&](int idx) { return (const float*)d_in[idx]; };
  const int ENC_CATE = 71, ENC_POI = 75, ENC_USER = 79;

  // ---- workspace bump allocator (256B aligned) ----
  char* wp = (char*)d_ws;
  auto alloc = [&](size_t bytes) -> void* {
    void* r = (void*)wp;
    wp += (bytes + 255) & ~(size_t)255;
    return r;
  };
  _Float16* hu_h = (_Float16*)alloc((size_t)N_USER * 64 * 2);
  _Float16* hp_h = (_Float16*)alloc((size_t)N_POI * 64 * 2);
  _Float16* hc_h = (_Float16*)alloc((size_t)N_CATE * 64 * 2);
  ConvBufs B;
  B.w16a  = (_Float16*)alloc((size_t)256 * 256 * 2);
  B.w16b  = (_Float16*)alloc((size_t)256 * 256 * 2);
  B.xl    = (float*)alloc((size_t)N_USER * 256 * 4);
  B.xr    = (float*)alloc((size_t)N_USER * 256 * 4);
  B.logit = (float*)alloc((size_t)E_UP * 4 * 4);
  B.segm  = (float*)alloc((size_t)N_USER * 4 * 4);
  B.denom = (float*)alloc((size_t)N_USER * 4 * 4);
  float* aggA = (float*)alloc((size_t)N_USER * 256 * 4);   // pu agg, then up agg, then L2
  float* aggB = (float*)alloc((size_t)N_POI * 256 * 4);    // pc agg, cp agg, then L2
  _Float16* user1_h = (_Float16*)alloc((size_t)N_USER * 256 * 2);
  _Float16* poi1_h  = (_Float16*)alloc((size_t)N_POI * 256 * 2);
  _Float16* cate1_h = (_Float16*)alloc((size_t)N_CATE * 256 * 2);

  // ---- encoders -> fp16 hidden states ----
  encode_kernel<<<ceil_div(N_USER * 64, 256), 256, 0, stream>>>(
      x_user, F(ENC_USER), F(ENC_USER + 1), F(ENC_USER + 3), F(ENC_USER + 2), hu_h, N_USER, 8);
  encode_kernel<<<ceil_div(N_POI * 64, 256), 256, 0, stream>>>(
      x_poi, F(ENC_POI), F(ENC_POI + 1), F(ENC_POI + 3), F(ENC_POI + 2), hp_h, N_POI, 9);
  encode_kernel<<<ceil_div(N_CATE * 64, 256), 256, 0, stream>>>(
      x_cate, F(ENC_CATE), F(ENC_CATE + 1), F(ENC_CATE + 3), F(ENC_CATE + 2), hc_h, N_CATE, 5);

  // ---- layer 1 (heads=4, d=256) ----
  run_conv(stream, B, hp_h, N_POI, hu_h, N_USER, 64, 4,
           F(PU1), F(PU1 + 1), F(PU1 + 2), F(PU1 + 3), F(PU1 + 5), F(PU1 + 6),
           ea_pu, src_pu, dst_pu, E_UP, aggA);
  finalize_relu_h<<<ceil_div(N_USER * 256, 256), 256, 0, stream>>>(aggA, F(PU1 + 4), user1_h, N_USER, 256);
  run_conv(stream, B, hp_h, N_POI, hc_h, N_CATE, 64, 4,
           F(PC1), F(PC1 + 1), F(PC1 + 2), F(PC1 + 3), F(PC1 + 5), F(PC1 + 6),
           ea_pc, src_pc, dst_pc, E_PC, aggB);
  finalize_relu_h<<<ceil_div(N_CATE * 256, 256), 256, 0, stream>>>(aggB, F(PC1 + 4), cate1_h, N_CATE, 256);
  run_conv(stream, B, hu_h, N_USER, hp_h, N_POI, 64, 4,
           F(UP1), F(UP1 + 1), F(UP1 + 2), F(UP1 + 3), F(UP1 + 5), F(UP1 + 6),
           ea_up, src_up, dst_up, E_UP, aggA);
  run_conv(stream, B, hc_h, N_CATE, hp_h, N_POI, 64, 4,
           F(CP1), F(CP1 + 1), F(CP1 + 2), F(CP1 + 3), F(CP1 + 5), F(CP1 + 6),
           ea_cp, src_cp, dst_cp, E_PC, aggB);
  finalize_relu_pair_h<<<ceil_div(N_POI * 256, 256), 256, 0, stream>>>(
      aggA, F(UP1 + 4), aggB, F(CP1 + 4), poi1_h, N_POI, 256);

  // ---- layer 2 (heads=1, d=64, K=256) ----
  float* user2 = (float*)d_out;
  float* poi2  = (float*)d_out + (size_t)N_USER * 64;
  float* cate2 = (float*)d_out + (size_t)(N_USER + N_POI) * 64;
  run_conv(stream, B, poi1_h, N_POI, user1_h, N_USER, 256, 1,
           F(PU2), F(PU2 + 1), F(PU2 + 2), F(PU2 + 3), F(PU2 + 5), F(PU2 + 6),
           ea_pu, src_pu, dst_pu, E_UP, aggA);
  out_single<<<ceil_div(N_USER * 64, 256), 256, 0, stream>>>(aggA, F(PU2 + 4), user2, N_USER, 64);
  run_conv(stream, B, poi1_h, N_POI, cate1_h, N_CATE, 256, 1,
           F(PC2), F(PC2 + 1), F(PC2 + 2), F(PC2 + 3), F(PC2 + 5), F(PC2 + 6),
           ea_pc, src_pc, dst_pc, E_PC, aggA);
  out_single<<<ceil_div(N_CATE * 64, 256), 256, 0, stream>>>(aggA, F(PC2 + 4), cate2, N_CATE, 64);
  run_conv(stream, B, user1_h, N_USER, poi1_h, N_POI, 256, 1,
           F(UP2), F(UP2 + 1), F(UP2 + 2), F(UP2 + 3), F(UP2 + 5), F(UP2 + 6),
           ea_up, src_up, dst_up, E_UP, aggA);
  run_conv(stream, B, cate1_h, N_CATE, poi1_h, N_POI, 256, 1,
           F(CP2), F(CP2 + 1), F(CP2 + 2), F(CP2 + 3), F(CP2 + 5), F(CP2 + 6),
           ea_cp, src_cp, dst_cp, E_PC, aggB);
  out_pair<<<ceil_div(N_POI * 64, 256), 256, 0, stream>>>(
      aggA, F(UP2 + 4), aggB, F(CP2 + 4), poi2, N_POI, 64);
}
